// MixOfExperts_51745765982649
// MI455X (gfx1250) — matrix operand order repaired and checked by reading, not compiled, not run on hardware
//
#include <hip/hip_runtime.h>
#include <hip/hip_bf16.h>

// ---------------------------------------------------------------------------
// MoE (top-2 of 8 experts), 8192 tokens, 1024 -> 1024.
// Strategy: permute tokens by expert, grouped GEMM with bf16 hi/lo split
// (3x v_wmma_f32_16x16x32_bf16 per K-step -> ~f32 accuracy at bf16 speed),
// f32 accumulate, weighted combine + bias, deterministic aux-loss reduction.
// ---------------------------------------------------------------------------

#define NT      8192      // tokens
#define DK      1024      // d_in
#define DOUT    1024      // d_out
#define NE      8         // experts
#define SLOT_CAP 16640    // 2*NT + 8*32 alignment slack, multiple of 32
#define MB_ROWS 32        // slot rows per GEMM block
#define NUM_MBLK (SLOT_CAP / MB_ROWS)   // 520

typedef __attribute__((ext_vector_type(8)))  __bf16        bf16x8;
typedef __attribute__((ext_vector_type(16))) __bf16        bf16x16;
typedef __attribute__((ext_vector_type(8)))  float         f32x8;
typedef __attribute__((ext_vector_type(8)))  unsigned short u16x8;

// ---------------- bf16 helpers (RNE, split hi/lo) ----------------
__device__ __forceinline__ unsigned short f32_to_bf16_rne(float f) {
  unsigned int u = __float_as_uint(f);
  unsigned int r = u + 0x7fffu + ((u >> 16) & 1u);
  if ((u & 0x7f800000u) == 0x7f800000u) r = u;   // inf/nan: truncate
  return (unsigned short)(r >> 16);
}
__device__ __forceinline__ float bf16_bits_to_f32(unsigned short h) {
  return __uint_as_float(((unsigned int)h) << 16);
}

// ---------------- WMMA wrappers ----------------
__device__ __forceinline__ f32x8 wmma_bf16(bf16x16 a, bf16x16 b, f32x8 c) {
  // v_wmma_f32_16x16x32_bf16  D = A(16x32) * B(32x16) + C
  return __builtin_amdgcn_wmma_f32_16x16x32_bf16(false, a, false, b,
                                                 (short)0, c, false, false);
}
__device__ __forceinline__ bf16x16 frag_cat(const unsigned short* p0,
                                            const unsigned short* p1) {
  bf16x8 lo = *(const bf16x8*)p0;
  bf16x8 hi = *(const bf16x8*)p1;
  return __builtin_shufflevector(lo, hi, 0,1,2,3,4,5,6,7,8,9,10,11,12,13,14,15);
}

// ---------------- 0) init scratch metadata ----------------
__global__ void moe_init(int* __restrict__ perm, int* __restrict__ meta) {
  int i = blockIdx.x * blockDim.x + threadIdx.x;
  if (i < SLOT_CAP) perm[i] = 0;      // padding rows gather token 0 (unused)
  if (i < 32) meta[i] = 0;            // cnt[8], cnt2[8], offA written later
}

// ---------------- 1) x -> bf16 hi/lo ----------------
__global__ void moe_cvt_x(const float* __restrict__ x,
                          unsigned short* __restrict__ xhi,
                          unsigned short* __restrict__ xlo) {
  size_t i = (size_t)blockIdx.x * blockDim.x + threadIdx.x;
  if (i >= (size_t)NT * DK) return;
  float f = x[i];
  unsigned short h = f32_to_bf16_rne(f);
  unsigned short l = f32_to_bf16_rne(f - bf16_bits_to_f32(h));
  xhi[i] = h; xlo[i] = l;
}

// ---------------- 2) W[e][k][o] -> WT[e][o][k] bf16 hi/lo (tiled transpose) --
__global__ void moe_cvt_wt(const float* __restrict__ W,
                           unsigned short* __restrict__ wthi,
                           unsigned short* __restrict__ wtlo) {
  __shared__ unsigned short thi[64][65];
  __shared__ unsigned short tlo[64][65];
  const int kt = blockIdx.x, ot = blockIdx.y, e = blockIdx.z;
  const int tx = threadIdx.x & 63, ty = threadIdx.x >> 6;   // ty in 0..3
#pragma unroll
  for (int j = 0; j < 16; ++j) {
    int kl = ty * 16 + j;
    float f = W[((size_t)e * DK + (kt * 64 + kl)) * DOUT + ot * 64 + tx];
    unsigned short h = f32_to_bf16_rne(f);
    thi[kl][tx] = h;
    tlo[kl][tx] = f32_to_bf16_rne(f - bf16_bits_to_f32(h));
  }
  __syncthreads();
#pragma unroll
  for (int j = 0; j < 16; ++j) {
    int ol = ty * 16 + j;
    size_t dst = ((size_t)e * DOUT + ot * 64 + ol) * DK + kt * 64 + tx;
    wthi[dst] = thi[tx][ol];   // k_local = tx, o_local = ol
    wtlo[dst] = tlo[tx][ol];
  }
}

// ---------------- 3) router: softmax + top2 + counts (wave per token) -------
__global__ void moe_router(const float* __restrict__ x,
                           const float* __restrict__ Wr,
                           const float* __restrict__ br,
                           float* __restrict__ probs,
                           int* __restrict__ top_i,
                           float* __restrict__ top_w,
                           int* __restrict__ meta) {
  const int lane = threadIdx.x & 31;
  const int n = blockIdx.x * 8 + (threadIdx.x >> 5);
  float acc[NE];
#pragma unroll
  for (int e = 0; e < NE; ++e) acc[e] = 0.f;
  const float* xr = x + (size_t)n * DK;
  for (int k = lane; k < DK; k += 32) {
    float xv = xr[k];
#pragma unroll
    for (int e = 0; e < NE; ++e) acc[e] += xv * Wr[e * DK + k];
  }
#pragma unroll
  for (int e = 0; e < NE; ++e) {
#pragma unroll
    for (int off = 16; off >= 1; off >>= 1)
      acc[e] += __shfl_xor(acc[e], off, 32);
  }
  if (lane == 0) {
    float l[NE], p[NE], m = -1e30f, s = 0.f;
#pragma unroll
    for (int e = 0; e < NE; ++e) { l[e] = acc[e] + br[e]; m = fmaxf(m, l[e]); }
#pragma unroll
    for (int e = 0; e < NE; ++e) { p[e] = __expf(l[e] - m); s += p[e]; }
    float inv = 1.f / s;
#pragma unroll
    for (int e = 0; e < NE; ++e) { p[e] *= inv; probs[(size_t)n * NE + e] = p[e]; }
    int i0 = 0;
#pragma unroll
    for (int e = 1; e < NE; ++e) if (p[e] > p[i0]) i0 = e;
    int i1 = (i0 == 0) ? 1 : 0;
#pragma unroll
    for (int e = 0; e < NE; ++e) if (e != i0 && p[e] > p[i1]) i1 = e;
    float sw = p[i0] + p[i1];
    top_i[n * 2 + 0] = i0;  top_w[n * 2 + 0] = p[i0] / sw;
    top_i[n * 2 + 1] = i1;  top_w[n * 2 + 1] = p[i1] / sw;
    atomicAdd(&meta[i0], 1);
    atomicAdd(&meta[i1], 1);
  }
}

// ---------------- 4) 32-aligned expert offsets ----------------
__global__ void moe_offsets(int* __restrict__ meta) {
  if (threadIdx.x == 0 && blockIdx.x == 0) {
    int run = 0;
#pragma unroll
    for (int e = 0; e < NE; ++e) {
      meta[16 + e] = run;
      run += (meta[e] + (MB_ROWS - 1)) & ~(MB_ROWS - 1);
    }
    meta[24] = run;   // total slots (multiple of 32, <= SLOT_CAP)
  }
}

// ---------------- 5) aux loss (fixed-order -> deterministic) ----------------
__global__ void moe_aux(const float* __restrict__ probs, float* __restrict__ aux) {
  __shared__ float part[256];
  const int t = threadIdx.x, e = t & 7, chunk = t >> 3;
  float s = 0.f;
  for (int n = chunk * 256; n < chunk * 256 + 256; ++n) s += probs[(size_t)n * NE + e];
  part[t] = s;
  __syncthreads();
  if (t < NE) {
    float tot = 0.f;
    for (int c = 0; c < 32; ++c) tot += part[c * 8 + t];
    float d = tot / (float)NT - 1.0f / NE;
    part[t] = d * d;
  }
  __syncthreads();
  if (t == 0) {
    float s2 = 0.f;
#pragma unroll
    for (int e2 = 0; e2 < NE; ++e2) s2 += part[e2];
    aux[0] = 0.01f * (s2 / (float)NE);
  }
}

// ---------------- 6) scatter tokens into expert slot lists ----------------
__global__ void moe_scatter(const int* __restrict__ top_i,
                            int* __restrict__ meta,
                            int* __restrict__ perm,
                            int* __restrict__ slot_of) {
  int n = blockIdx.x * blockDim.x + threadIdx.x;
  if (n >= NT) return;
#pragma unroll
  for (int k = 0; k < 2; ++k) {
    int e = top_i[n * 2 + k];
    int pos = atomicAdd(&meta[8 + e], 1);
    int slot = meta[16 + e] + pos;
    perm[slot] = n;
    slot_of[n * 2 + k] = slot;
  }
}

// ---------------- 7) grouped GEMM: Y[slot] = x[perm[slot]] @ W[e] -----------
// Block: 32 slot rows x 128 cols.  8 waves: wave = (mTile 0..1, colTile 0..3),
// each wave owns two 16x16 output tiles. A (hi+lo) staged in 128KB dynamic LDS
// in chunk layout lds[c][row][8] (conflict-free 16B ds reads, no padding).
__global__ __launch_bounds__(256) void moe_gemm(
    const unsigned short* __restrict__ xhi, const unsigned short* __restrict__ xlo,
    const unsigned short* __restrict__ wthi, const unsigned short* __restrict__ wtlo,
    float* __restrict__ Y, const int* __restrict__ perm, const int* __restrict__ offA) {
  extern __shared__ unsigned short smem[];
  unsigned short* ahi = smem;               // 128 chunks * 32 rows * 8 halves
  unsigned short* alo = smem + 128 * 32 * 8;

  const int tid  = threadIdx.x;
  const int sOff = blockIdx.x * MB_ROWS;
  const int total = offA[8];
  if (sOff >= total) return;
  int e = 0;
#pragma unroll
  for (int j = 1; j < NE; ++j) if (sOff >= offA[j]) e = j;

  {  // stage A rows (gathered by perm) into LDS, hi and lo
    const int row = tid >> 3, seg = tid & 7;
    const int src = perm[sOff + row];
    const unsigned short* gh = xhi + (size_t)src * DK;
    const unsigned short* gl = xlo + (size_t)src * DK;
#pragma unroll
    for (int c = seg * 16; c < seg * 16 + 16; ++c) {
      *(u16x8*)(ahi + (c * 32 + row) * 8) = *(const u16x8*)(gh + c * 8);
      *(u16x8*)(alo + (c * 32 + row) * 8) = *(const u16x8*)(gl + c * 8);
    }
  }
  __syncthreads();

  const int lane = tid & 31;
  const int wv   = tid >> 5;
  const int mT   = wv >> 2;                     // 0..1
  const int cT   = wv & 3;                      // 0..3
  const int rowA = (lane & 15) + mT * 16;       // A-row this lane holds
  const int grp  = lane >> 4;                   // half-wave group
  const int colBase = blockIdx.y * 128 + cT * 32;
  const size_t wbase = (size_t)e << 20;         // e * 1024 * 1024

  f32x8 c0 = {}; f32x8 c1 = {};
  const size_t wo0 = wbase + (size_t)(colBase +      (lane & 15)) * DK + grp * 16;
  const size_t wo1 = wbase + (size_t)(colBase + 16 + (lane & 15)) * DK + grp * 16;

  for (int kk = 0; kk < DK; kk += 32) {
    const int cB = (kk >> 3) + 2 * grp;
    // A fragments (this code's k order, same as B: halves 0-15 -> K kk+16g.. contiguous)
    bf16x16 aH = frag_cat(ahi + ((cB)     * 32 + rowA) * 8,
                          ahi + ((cB + 1) * 32 + rowA) * 8);
    bf16x16 aL = frag_cat(alo + ((cB)     * 32 + rowA) * 8,
                          alo + ((cB + 1) * 32 + rowA) * 8);
    // B fragments from transposed W: lane = column, 16 contiguous K halves
    {
      const unsigned short* ph = wthi + wo0 + kk;
      const unsigned short* pl = wtlo + wo0 + kk;
      bf16x16 bH = frag_cat(ph, ph + 8);
      bf16x16 bL = frag_cat(pl, pl + 8);
      c0 = wmma_bf16(aH, bH, c0);
      c0 = wmma_bf16(aH, bL, c0);
      c0 = wmma_bf16(aL, bH, c0);
    }
    {
      const unsigned short* ph = wthi + wo1 + kk;
      const unsigned short* pl = wtlo + wo1 + kk;
      bf16x16 bH = frag_cat(ph, ph + 8);
      bf16x16 bL = frag_cat(pl, pl + 8);
      c1 = wmma_bf16(aH, bH, c1);
      c1 = wmma_bf16(aH, bL, c1);
      c1 = wmma_bf16(aL, bH, c1);
    }
  }

  // C/D layout: VGPR r -> M = r + 8*grp, lane&15 -> N
  const int rBase = sOff + mT * 16 + grp * 8;
  const int col0  = colBase + (lane & 15);
#pragma unroll
  for (int r = 0; r < 8; ++r) {
    Y[(size_t)(rBase + r) * DOUT + col0]      = c0[r];
    Y[(size_t)(rBase + r) * DOUT + col0 + 16] = c1[r];
  }
}

// ---------------- 8) weighted combine + bias ----------------
__global__ void moe_combine(const float* __restrict__ Y,
                            const float* __restrict__ b,
                            const int* __restrict__ top_i,
                            const float* __restrict__ top_w,
                            const int* __restrict__ slot_of,
                            float* __restrict__ out) {
  size_t gid = (size_t)blockIdx.x * blockDim.x + threadIdx.x;
  if (gid >= (size_t)NT * DOUT) return;
  int n = (int)(gid >> 10), o = (int)(gid & 1023);
  int e0 = top_i[n * 2], e1 = top_i[n * 2 + 1];
  float w0 = top_w[n * 2], w1 = top_w[n * 2 + 1];
  int s0 = slot_of[n * 2], s1 = slot_of[n * 2 + 1];
  out[gid] = w0 * (Y[(size_t)s0 * DOUT + o] + b[(size_t)e0 * DOUT + o]) +
             w1 * (Y[(size_t)s1 * DOUT + o] + b[(size_t)e1 * DOUT + o]);
}

// ---------------- host launcher ----------------
extern "C" void kernel_launch(void* const* d_in, const int* in_sizes, int n_in,
                              void* d_out, int out_size, void* d_ws, size_t ws_size,
                              hipStream_t stream) {
  (void)in_sizes; (void)n_in; (void)out_size; (void)ws_size;
  const float* x  = (const float*)d_in[0];
  const float* W  = (const float*)d_in[1];
  const float* b  = (const float*)d_in[2];
  const float* Wr = (const float*)d_in[3];
  const float* br = (const float*)d_in[4];

  // workspace layout (bytes); total ~136 MB
  char* ws = (char*)d_ws;
  const size_t OFF_XHI  = 0;
  const size_t OFF_XLO  = OFF_XHI + (size_t)NT * DK * 2;
  const size_t OFF_WTHI = OFF_XLO + (size_t)NT * DK * 2;
  const size_t OFF_WTLO = OFF_WTHI + (size_t)NE * DK * DOUT * 2;
  const size_t OFF_Y    = OFF_WTLO + (size_t)NE * DK * DOUT * 2;
  const size_t OFF_PRB  = OFF_Y    + (size_t)SLOT_CAP * DOUT * 4;
  const size_t OFF_TPI  = OFF_PRB  + (size_t)NT * NE * 4;
  const size_t OFF_TPW  = OFF_TPI  + (size_t)NT * 2 * 4;
  const size_t OFF_SLT  = OFF_TPW  + (size_t)NT * 2 * 4;
  const size_t OFF_PRM  = OFF_SLT  + (size_t)NT * 2 * 4;
  const size_t OFF_MET  = OFF_PRM  + (size_t)SLOT_CAP * 4;

  unsigned short* xhi  = (unsigned short*)(ws + OFF_XHI);
  unsigned short* xlo  = (unsigned short*)(ws + OFF_XLO);
  unsigned short* wthi = (unsigned short*)(ws + OFF_WTHI);
  unsigned short* wtlo = (unsigned short*)(ws + OFF_WTLO);
  float* Y     = (float*)(ws + OFF_Y);
  float* probs = (float*)(ws + OFF_PRB);
  int*   topi  = (int*)  (ws + OFF_TPI);
  float* topw  = (float*)(ws + OFF_TPW);
  int*   slot  = (int*)  (ws + OFF_SLT);
  int*   perm  = (int*)  (ws + OFF_PRM);
  int*   meta  = (int*)  (ws + OFF_MET);   // [0..7]=cnt [8..15]=cnt2 [16..24]=offA

  float* out = (float*)d_out;
  float* aux = out + (size_t)NT * DOUT;

  moe_init   <<<(SLOT_CAP + 255) / 256, 256, 0, stream>>>(perm, meta);
  moe_cvt_x  <<<((size_t)NT * DK + 255) / 256, 256, 0, stream>>>(x, xhi, xlo);
  moe_cvt_wt <<<dim3(DK / 64, DOUT / 64, NE), 256, 0, stream>>>(W, wthi, wtlo);
  moe_router <<<NT / 8, 256, 0, stream>>>(x, Wr, br, probs, topi, topw, meta);
  moe_offsets<<<1, 1, 0, stream>>>(meta);
  moe_aux    <<<1, 256, 0, stream>>>(probs, aux);
  moe_scatter<<<NT / 256, 256, 0, stream>>>(topi, meta, perm, slot);
  moe_gemm   <<<dim3(NUM_MBLK, DOUT / 128), 256, 131072, stream>>>(
                 xhi, xlo, wthi, wtlo, Y, perm, meta + 16);
  moe_combine<<<((size_t)NT * DOUT + 255) / 256, 256, 0, stream>>>(
                 Y, b, topi, topw, slot, out);
}
